// TrXLI_73100343378119
// MI455X (gfx1250) — matrix operand order repaired and checked by reading, not compiled, not run on hardware
//
#include <hip/hip_runtime.h>
#include <hip/hip_bf16.h>
#include <cstdint>

// TrXL-I block, rank-1-query optimized pipeline for gfx1250 (wave32, WMMA).
// K/V projections are algebraically collapsed (q-len == 1) -> HBM-bound;
// all matmuls use v_wmma_f32_16x16x32_f16 with 16x64 (or 32x16) per-wave
// tiles for operand reuse; attention operands are staged to LDS with
// global_load_async_to_lds_b128 (ASYNCcnt double buffering).

typedef __attribute__((ext_vector_type(16))) _Float16 v16h;
typedef __attribute__((ext_vector_type(8)))  _Float16 v8h;
typedef __attribute__((ext_vector_type(4)))  _Float16 v4h;
typedef __attribute__((ext_vector_type(8)))  float    v8f;

#define D_  1024
#define H_  16
#define DH_ 64
#define B_  64
#define L_  512

// ---------------- WMMA helpers (layouts per CDNA5 ISA) ----------

__device__ __forceinline__ v8f wmma_f16(v16h a, v16h b, v8f c) {
  // 8 args: (neg_a, A, neg_b, B, c_mod, C, reuse_a, reuse_b)
  return __builtin_amdgcn_wmma_f32_16x16x32_f16(false, a, false, b, (short)0, c,
                                                false, false);
}

// A fragment (16x32 f16): lane row = lane&15, g = lane>>4.
// halves[0..7]  = A[row][kk + g*8 + 0..7]
// halves[8..15] = A[row][kk + 16 + g*8 + 0..7]
__device__ __forceinline__ v16h load_afrag(const _Float16* A, int lda, int row,
                                           int kk, int g) {
  const _Float16* p = A + (size_t)row * lda + kk + g * 8;
  v8h lo = *(const v8h*)p;
  v8h hi = *(const v8h*)(p + 16);
  return __builtin_shufflevector(lo, hi, 0, 1, 2, 3, 4, 5, 6, 7,
                                 8, 9, 10, 11, 12, 13, 14, 15);
}

// B fragment (32x16 f16) from transposed storage BT[n][k]:
// lane col n = lane&15, halves[0..7] = BT[n][kk + g*8 + 0..7],
// halves[8..15] = BT[n][kk + 16 + g*8 + 0..7]
__device__ __forceinline__ v16h load_bfrag(const _Float16* BT, int ldb, int col,
                                           int kk, int g) {
  const _Float16* p = BT + (size_t)col * ldb + kk + g * 8;
  v8h lo = *(const v8h*)p;
  v8h hi = *(const v8h*)(p + 16);
  return __builtin_shufflevector(lo, hi, 0, 1, 2, 3, 4, 5, 6, 7,
                                 8, 9, 10, 11, 12, 13, 14, 15);
}

// ---------------- Async global->LDS (CDNA5, tracked by ASYNCcnt) -----------
// Generic pointers into LDS have addr[31:0] == wave-relative LDS address
// (ISA 10.2 aperture rules), so truncation gives the VDST operand.
__device__ __forceinline__ void async_g2l_b128(uint32_t lds_byte_addr,
                                               const _Float16* gp) {
  asm volatile("global_load_async_to_lds_b128 %0, %1, off"
               :: "v"(lds_byte_addr), "v"(gp) : "memory");
}
__device__ __forceinline__ uint32_t lds_addr(const void* p) {
  return (uint32_t)(uintptr_t)p;
}

// ---------------- Generic WMMA GEMM: C = A[MxK] * BT[NxK]^T ----------------
// 16x64 output stripe per wave. All 5 fragments are loaded into distinct
// registers before the 4 WMMAs so loads clause together and the waits
// stagger against the WMMA pipe. Optional batching over blockIdx.y.
__global__ void __launch_bounds__(256) gemm16(
    const _Float16* __restrict__ A, int lda, long long aBS,
    const _Float16* __restrict__ BT, int ldb, long long bBS,
    const float* __restrict__ bias, long long biasBS,
    const float* __restrict__ res, int ldres,
    float* __restrict__ outF, _Float16* __restrict__ outH, int ldc, long long oBS,
    _Float16* __restrict__ outHT, int ldct, long long otBS,
    int M, int N, int K, float scale, int relu)
{
  const long long batch = blockIdx.y;
  A  += batch * aBS;
  BT += batch * bBS;

  const int wave = blockIdx.x * (blockDim.x >> 5) + (threadIdx.x >> 5);
  const int lane = threadIdx.x & 31;
  const int nstripes = N >> 6;                 // stripes of 64 columns
  const int total    = (M >> 4) * nstripes;
  if (wave >= total) return;
  const int mt = wave / nstripes;
  const int ns = wave - mt * nstripes;
  const int g  = lane >> 4;
  const int nl = lane & 15;
  const int row  = mt * 16 + nl;
  const int col0 = ns * 64 + nl;

  v8f c0 = {}, c1 = {}, c2 = {}, c3 = {};
  for (int kk = 0; kk < K; kk += 32) {
    v16h a  = load_afrag(A, lda, row, kk, g);
    v16h b0 = load_bfrag(BT, ldb, col0,      kk, g);
    v16h b1 = load_bfrag(BT, ldb, col0 + 16, kk, g);
    v16h b2 = load_bfrag(BT, ldb, col0 + 32, kk, g);
    v16h b3 = load_bfrag(BT, ldb, col0 + 48, kk, g);
    c0 = wmma_f16(a, b0, c0);
    c1 = wmma_f16(a, b1, c1);
    c2 = wmma_f16(a, b2, c2);
    c3 = wmma_f16(a, b3, c3);
  }

  const v8f cs[4] = {c0, c1, c2, c3};
#pragma unroll
  for (int j = 0; j < 4; ++j) {
    const int n = ns * 64 + j * 16 + nl;
    const float bvv = bias ? bias[batch * biasBS + n] : 0.f;
#pragma unroll
    for (int r = 0; r < 8; ++r) {
      const int m = mt * 16 + g * 8 + r;  // C layout: lanes 0-15 rows 0-7, 16-31 rows 8-15
      float vv = cs[j][r] * scale + bvv;
      if (relu) vv = fmaxf(vv, 0.f);
      if (res)  vv += res[(size_t)m * ldres + n];
      if (outF)  outF [batch * oBS  + (size_t)m * ldc  + n] = vv;
      if (outH)  outH [batch * oBS  + (size_t)m * ldc  + n] = (_Float16)vv;
      if (outHT) outHT[batch * otBS + (size_t)n * ldct + m] = (_Float16)vv;
    }
  }
}

// ---------------- LayerNorm over rows of width 1024 ------------------------
__global__ void __launch_bounds__(256) ln_rows(
    const float* __restrict__ in, const float* __restrict__ gam,
    const float* __restrict__ bet, _Float16* __restrict__ outH,
    float* __restrict__ outF)
{
  __shared__ float s1[256], s2[256];
  const int row = blockIdx.x, tid = threadIdx.x;
  const float4 x = ((const float4*)(in + (size_t)row * D_))[tid];
  s1[tid] = x.x + x.y + x.z + x.w;
  s2[tid] = x.x * x.x + x.y * x.y + x.z * x.z + x.w * x.w;
  __syncthreads();
  for (int s = 128; s > 0; s >>= 1) {
    if (tid < s) { s1[tid] += s1[tid + s]; s2[tid] += s2[tid + s]; }
    __syncthreads();
  }
  const float mu   = s1[0] * (1.f / D_);
  const float var  = s2[0] * (1.f / D_) - mu * mu;
  const float rstd = rsqrtf(var + 1e-5f);
  const float4 gg = ((const float4*)gam)[tid];
  const float4 bb = ((const float4*)bet)[tid];
  const float o0 = (x.x - mu) * rstd * gg.x + bb.x;
  const float o1 = (x.y - mu) * rstd * gg.y + bb.y;
  const float o2 = (x.z - mu) * rstd * gg.z + bb.z;
  const float o3 = (x.w - mu) * rstd * gg.w + bb.w;
  if (outF) ((float4*)(outF + (size_t)row * D_))[tid] = make_float4(o0, o1, o2, o3);
  v4h h = {(_Float16)o0, (_Float16)o1, (_Float16)o2, (_Float16)o3};
  ((v4h*)(outH + (size_t)row * D_))[tid] = h;
}

// Per-row mean / rstd only (for the fused LN+transpose of v).
__global__ void __launch_bounds__(256) row_stats(const float* __restrict__ in,
                                                 float* __restrict__ stats)
{
  __shared__ float s1[256], s2[256];
  const int row = blockIdx.x, tid = threadIdx.x;
  const float4 x = ((const float4*)(in + (size_t)row * D_))[tid];
  s1[tid] = x.x + x.y + x.z + x.w;
  s2[tid] = x.x * x.x + x.y * x.y + x.z * x.z + x.w * x.w;
  __syncthreads();
  for (int s = 128; s > 0; s >>= 1) {
    if (tid < s) { s1[tid] += s1[tid + s]; s2[tid] += s2[tid + s]; }
    __syncthreads();
  }
  if (tid == 0) {
    const float mu  = s1[0] * (1.f / D_);
    const float var = s2[0] * (1.f / D_) - mu * mu;
    stats[row * 2]     = mu;
    stats[row * 2 + 1] = rsqrtf(var + 1e-5f);
  }
}

// LN(v) fused with transpose: vnT[b][d][l] (f16), 32x32 tiles through LDS.
__global__ void __launch_bounds__(256) ln_transpose_v(
    const float* __restrict__ v, const float* __restrict__ stats,
    const float* __restrict__ gam, const float* __restrict__ bet,
    _Float16* __restrict__ vnT)
{
  __shared__ _Float16 tile[32 * 36];  // [d_local][l_local], stride 36 for 8B-aligned v4h reads
  const int bid = blockIdx.x;
  const int b  = bid >> 9;          // 16*32 tiles per batch
  const int lt = (bid >> 5) & 15;
  const int dt = bid & 31;
  const int tid = threadIdx.x;
  const int r = tid >> 3;           // 0..31
  const int c = (tid & 7) * 4;      // 0..28
  const int l = lt * 32 + r;
  const int d0 = dt * 32 + c;
  const float4 x = *(const float4*)(v + ((size_t)b * L_ + l) * D_ + d0);
  const float mu   = stats[((size_t)b * L_ + l) * 2];
  const float rstd = stats[((size_t)b * L_ + l) * 2 + 1];
  const float xv[4] = {x.x, x.y, x.z, x.w};
#pragma unroll
  for (int i = 0; i < 4; ++i) {
    const float o = (xv[i] - mu) * rstd * gam[d0 + i] + bet[d0 + i];
    tile[(c + i) * 36 + r] = (_Float16)o;
  }
  __syncthreads();
  const v4h h = *(const v4h*)&tile[r * 36 + c];
  *(v4h*)(vnT + ((size_t)b * D_ + dt * 32 + r) * L_ + lt * 32 + c) = h;
}

// f32 [1024x1024] -> transposed f16 [1024x1024]
__global__ void __launch_bounds__(256) transpose_cvt(const float* __restrict__ in,
                                                     _Float16* __restrict__ out)
{
  __shared__ _Float16 tile[32 * 36];
  const int bid = blockIdx.x;       // 32*32 tiles
  const int rt = bid >> 5, ct = bid & 31;
  const int tid = threadIdx.x;
  const int r = tid >> 3;
  const int c = (tid & 7) * 4;
  const float4 x = *(const float4*)(in + (size_t)(rt * 32 + r) * D_ + ct * 32 + c);
  tile[(c + 0) * 36 + r] = (_Float16)x.x;
  tile[(c + 1) * 36 + r] = (_Float16)x.y;
  tile[(c + 2) * 36 + r] = (_Float16)x.z;
  tile[(c + 3) * 36 + r] = (_Float16)x.w;
  __syncthreads();
  const v4h h = *(const v4h*)&tile[r * 36 + c];
  *(v4h*)(out + (size_t)(ct * 32 + r) * D_ + rt * 32 + c) = h;
}

// plain f32 -> f16 convert (same layout)
__global__ void __launch_bounds__(256) cvt_f16(const float* __restrict__ in,
                                               _Float16* __restrict__ out)
{
  const size_t i = (size_t)blockIdx.x * 256 + threadIdx.x;
  const float4 x = ((const float4*)in)[i];
  v4h h = {(_Float16)x.x, (_Float16)x.y, (_Float16)x.z, (_Float16)x.w};
  ((v4h*)out)[i] = h;
}

// ---------------- Attention, split into 3 kernels for occupancy ------------

// logits[b][h][l] = kn[b] @ Weff[b]  (scale pre-folded into Weff)
// grid (B, 4): each WG covers 128 l-rows; wave w computes 2 m-tiles (rows
// w*32..w*32+31) sharing the B fragments. kn is double-buffered into LDS in
// 128x64-half chunks via global_load_async_to_lds_b128 (4 WMMAs per barrier
// pair, next chunk DMA overlapped with current chunk math).
__global__ void __launch_bounds__(256) attn_logits(
    const _Float16* __restrict__ kn,    // [B*L][D]
    const _Float16* __restrict__ WeffT, // [B][H][D]
    float* __restrict__ logitsG)        // [B][H][L]
{
  __shared__ _Float16 kbuf[2][128 * 64];   // 2 x 16 KB: [l_local][k_chunk 64]
  const int b   = blockIdx.x;
  const int grp = blockIdx.y;              // 0..3
  const int tid = threadIdx.x;
  const int wave = tid >> 5, lane = tid & 31;
  const int g = lane >> 4, nl = lane & 15;
  const int l0 = grp * 128;

  const _Float16* knB = kn + ((size_t)b * L_ + l0) * D_;
  const _Float16* weB = WeffT + (size_t)b * H_ * D_;

  const int sr = tid >> 1;                 // staging row 0..127
  const int sc = (tid & 1) * 32;           // staging col (halves): 0 or 32
  auto stage = [&](int buf, int kk) {
    const _Float16* gp = knB + (size_t)sr * D_ + kk + sc;
    const uint32_t lo = lds_addr(&kbuf[buf][sr * 64 + sc]);
    async_g2l_b128(lo,      gp);
    async_g2l_b128(lo + 16, gp + 8);
    async_g2l_b128(lo + 32, gp + 16);
    async_g2l_b128(lo + 48, gp + 24);
  };

  stage(0, 0);                             // prologue: 4 async ops/wave in flight
  v8f acc0 = {}, acc1 = {};
  const int r0 = wave * 32;                // wave w -> l-rows l0+w*32 .. +31
  for (int kk = 0; kk < D_; kk += 64) {
    const int cur = (kk >> 6) & 1;
    if (kk + 64 < D_) {
      stage(cur ^ 1, kk + 64);             // prefetch next chunk (4 more ops)
      asm volatile("s_wait_asynccnt 0x4" ::: "memory");  // oldest 4 (cur) done
    } else {
      asm volatile("s_wait_asynccnt 0x0" ::: "memory");
    }
    __syncthreads();
    // Hoist all fragments, then issue 4 WMMAs back-to-back.
    v16h bb0 = load_bfrag(weB, D_, nl, kk,      g);   // 16 cols = heads
    v16h bb1 = load_bfrag(weB, D_, nl, kk + 32, g);
    v16h a00 = load_afrag(&kbuf[cur][0], 64, r0 + nl,      0,  g);
    v16h a10 = load_afrag(&kbuf[cur][0], 64, r0 + 16 + nl, 0,  g);
    v16h a01 = load_afrag(&kbuf[cur][0], 64, r0 + nl,      32, g);
    v16h a11 = load_afrag(&kbuf[cur][0], 64, r0 + 16 + nl, 32, g);
    acc0 = wmma_f16(a00, bb0, acc0);
    acc1 = wmma_f16(a10, bb0, acc1);
    acc0 = wmma_f16(a01, bb1, acc0);
    acc1 = wmma_f16(a11, bb1, acc1);
    __syncthreads();                       // cur buffer fully consumed
  }
  const v8f accs[2] = {acc0, acc1};
#pragma unroll
  for (int i = 0; i < 2; ++i)
#pragma unroll
    for (int r = 0; r < 8; ++r) {
      const int l = l0 + r0 + i * 16 + g * 8 + r;
      logitsG[((size_t)b * H_ + nl) * L_ + l] = accs[i][r];
    }
}

// softmax over L per (b,h); writes normalized att as f16 [b][h][l]
__global__ void __launch_bounds__(256) attn_softmax(
    const float* __restrict__ logitsG, _Float16* __restrict__ attG)
{
  __shared__ float red[256];
  const int bh = blockIdx.x;
  const int tid = threadIdx.x;
  const float x0 = logitsG[(size_t)bh * L_ + tid];
  const float x1 = logitsG[(size_t)bh * L_ + tid + 256];
  red[tid] = fmaxf(x0, x1);
  __syncthreads();
  for (int s = 128; s > 0; s >>= 1) {
    if (tid < s) red[tid] = fmaxf(red[tid], red[tid + s]);
    __syncthreads();
  }
  const float mx = red[0];
  __syncthreads();
  const float e0 = __expf(x0 - mx), e1 = __expf(x1 - mx);
  red[tid] = e0 + e1;
  __syncthreads();
  for (int s = 128; s > 0; s >>= 1) {
    if (tid < s) red[tid] += red[tid + s];
    __syncthreads();
  }
  const float inv = 1.f / red[0];
  attG[(size_t)bh * L_ + tid]       = (_Float16)(e0 * inv);
  attG[(size_t)bh * L_ + tid + 256] = (_Float16)(e1 * inv);
}

// Tv[b][h][d] = att(16xL) @ vn(LxD); grid (B, 2); wave w -> 16x64 d-stripe
// stripe = grp*8+w. att block (16 KB) staged via async-to-LDS; A fragment
// reused across 4 WMMAs; B-frags contiguous via vnT, hoisted for clausing.
__global__ void __launch_bounds__(256) attn_tv(
    const _Float16* __restrict__ attG,  // [B][H][L]
    const _Float16* __restrict__ vnT,   // [B][D][L]
    _Float16* __restrict__ Tv)          // [B][H][D]
{
  __shared__ _Float16 abuf[H_ * L_];    // 16 KB
  const int b   = blockIdx.x;
  const int grp = blockIdx.y;           // 0..1
  const int tid = threadIdx.x;
  const int wave = tid >> 5, lane = tid & 31;
  const int g = lane >> 4, nl = lane & 15;

  {
    const _Float16* ga = attG + (size_t)b * H_ * L_;
#pragma unroll
    for (int i = 0; i < 4; ++i) {
      const int idx = tid + i * 256;    // 16B chunk id, 1024 total
      async_g2l_b128(lds_addr(&abuf[idx * 8]), ga + idx * 8);
    }
    asm volatile("s_wait_asynccnt 0x0" ::: "memory");
    __syncthreads();
  }

  const _Float16* vb = vnT + (size_t)b * D_ * L_;
  const int d0 = (grp * 8 + wave) * 64;
  v8f c0 = {}, c1 = {}, c2 = {}, c3 = {};
  for (int kk = 0; kk < L_; kk += 32) {
    v16h a  = load_afrag(abuf, L_, nl, kk, g);          // A from LDS
    v16h b0 = load_bfrag(vb, L_, d0 + nl,      kk, g);  // contiguous via vnT
    v16h b1 = load_bfrag(vb, L_, d0 + 16 + nl, kk, g);
    v16h b2 = load_bfrag(vb, L_, d0 + 32 + nl, kk, g);
    v16h b3 = load_bfrag(vb, L_, d0 + 48 + nl, kk, g);
    c0 = wmma_f16(a, b0, c0);
    c1 = wmma_f16(a, b1, c1);
    c2 = wmma_f16(a, b2, c2);
    c3 = wmma_f16(a, b3, c3);
  }
  const v8f cs[4] = {c0, c1, c2, c3};
#pragma unroll
  for (int j = 0; j < 4; ++j)
#pragma unroll
    for (int r = 0; r < 8; ++r)
      Tv[(size_t)b * H_ * D_ + (size_t)(g * 8 + r) * D_ + d0 + j * 16 + nl] =
          (_Float16)cs[j][r];
}

// ---------------- Host-side pipeline ---------------------------------------
extern "C" void kernel_launch(void* const* d_in, const int* in_sizes, int n_in,
                              void* d_out, int out_size, void* d_ws, size_t ws_size,
                              hipStream_t stream)
{
  (void)in_sizes; (void)n_in; (void)out_size; (void)ws_size;
  const float* q       = (const float*)d_in[0];
  const float* k       = (const float*)d_in[1];
  const float* v       = (const float*)d_in[2];
  const float* ln_q_g  = (const float*)d_in[3];
  const float* ln_q_b  = (const float*)d_in[4];
  const float* ln_k_g  = (const float*)d_in[5];
  const float* ln_k_b  = (const float*)d_in[6];
  const float* ln_v_g  = (const float*)d_in[7];
  const float* ln_v_b  = (const float*)d_in[8];
  const float* ln_fc_g = (const float*)d_in[9];
  const float* ln_fc_b = (const float*)d_in[10];
  const float* Wq      = (const float*)d_in[11];
  const float* bq      = (const float*)d_in[12];
  const float* Wk      = (const float*)d_in[13];
  /* bk (d_in[14]) provably cancels in softmax */
  const float* Wv      = (const float*)d_in[15];
  const float* bv      = (const float*)d_in[16];
  const float* Wmh     = (const float*)d_in[17];
  const float* bmh     = (const float*)d_in[18];
  const float* Wfc     = (const float*)d_in[19];
  const float* bfc     = (const float*)d_in[20];

  char* ws = (char*)d_ws;
  size_t off = 0;
  auto alloc = [&](size_t bytes) -> char* {
    char* p = ws + off;
    off += (bytes + 255) & ~(size_t)255;
    return p;
  };
  float*    qn_f   = (float*)   alloc((size_t)B_ * D_ * 4);
  _Float16* qn_h   = (_Float16*)alloc((size_t)B_ * D_ * 2);
  _Float16* kn_h   = (_Float16*)alloc((size_t)B_ * L_ * D_ * 2);   // 64 MB
  float*    vst    = (float*)   alloc((size_t)B_ * L_ * 2 * 4);
  _Float16* vnT    = (_Float16*)alloc((size_t)B_ * L_ * D_ * 2);   // 64 MB
  _Float16* Wk_h   = (_Float16*)alloc((size_t)D_ * D_ * 2);
  _Float16* WqT    = (_Float16*)alloc((size_t)D_ * D_ * 2);
  _Float16* WvT    = (_Float16*)alloc((size_t)D_ * D_ * 2);
  _Float16* WmhT   = (_Float16*)alloc((size_t)D_ * D_ * 2);
  _Float16* WfcT   = (_Float16*)alloc((size_t)D_ * D_ * 2);
  _Float16* hq_h   = (_Float16*)alloc((size_t)B_ * D_ * 2);
  _Float16* WeffT  = (_Float16*)alloc((size_t)B_ * H_ * D_ * 2);
  float*    logitsG= (float*)   alloc((size_t)B_ * H_ * L_ * 4);   // 2 MB
  _Float16* attG   = (_Float16*)alloc((size_t)B_ * H_ * L_ * 2);   // 1 MB
  _Float16* Tv     = (_Float16*)alloc((size_t)B_ * H_ * D_ * 2);
  _Float16* ctx_h  = (_Float16*)alloc((size_t)B_ * D_ * 2);
  float*    hhead  = (float*)   alloc((size_t)B_ * D_ * 4);
  _Float16* hn_h   = (_Float16*)alloc((size_t)B_ * D_ * 2);

  // LayerNorms (+f16 casts), LN+transpose for v
  ln_rows<<<B_, 256, 0, stream>>>(q, ln_q_g, ln_q_b, qn_h, qn_f);
  ln_rows<<<B_ * L_, 256, 0, stream>>>(k, ln_k_g, ln_k_b, kn_h, nullptr);
  row_stats<<<B_ * L_, 256, 0, stream>>>(v, vst);
  ln_transpose_v<<<B_ * 16 * 32, 256, 0, stream>>>(v, vst, ln_v_g, ln_v_b, vnT);

  // Weight conversions (B-side operands stored transposed for WMMA layout)
  cvt_f16<<<1024, 256, 0, stream>>>(Wk, Wk_h);
  transpose_cvt<<<1024, 256, 0, stream>>>(Wq, WqT);
  transpose_cvt<<<1024, 256, 0, stream>>>(Wv, WvT);
  transpose_cvt<<<1024, 256, 0, stream>>>(Wmh, WmhT);
  transpose_cvt<<<1024, 256, 0, stream>>>(Wfc, WfcT);

  // hq = LN(q) @ Wq + bq    [64x1024x1024] -> f16; 4*16 = 64 stripes
  gemm16<<<dim3(8, 1), 256, 0, stream>>>(qn_h, D_, 0, WqT, D_, 0,
      bq, 0, nullptr, 0, nullptr, hq_h, D_, 0, nullptr, 0, 0,
      B_, D_, D_, 1.f, 0);

  // Weff[b,h,:] = scale * Wk[:,h*64:+64] @ hq[b,h*64:+64]  (per-head batched)
  // M=1024, N=64 -> 64 stripes per head
  gemm16<<<dim3(8, H_), 256, 0, stream>>>(Wk_h, D_, DH_, hq_h, D_, DH_,
      nullptr, 0, nullptr, 0, nullptr, nullptr, 0, 0,
      WeffT, H_ * D_, D_, D_, B_, DH_, 0.125f, 0);

  // attention: logits -> softmax -> Tv = att @ vn
  attn_logits<<<dim3(B_, 4), 256, 0, stream>>>(kn_h, WeffT, logitsG);
  attn_softmax<<<B_ * H_, 256, 0, stream>>>(logitsG, attG);
  attn_tv<<<dim3(B_, 2), 256, 0, stream>>>(attG, vnT, Tv);

  // ctx[:, h*64:+64] = Tv[:,h,:] @ Wv[:,h*64:+64] + bv  (per-head batched)
  // M=64, N=64 -> 4 stripes per head (waves 4..7 exit early)
  gemm16<<<dim3(1, H_), 256, 0, stream>>>(Tv, H_ * D_, D_,
      WvT, D_, (long long)DH_ * D_, bv, DH_, nullptr, 0,
      nullptr, ctx_h, D_, DH_, nullptr, 0, 0,
      B_, DH_, D_, 1.f, 0);

  // h_head = qn + ctx @ W_mhdpa + b_mhdpa
  gemm16<<<dim3(8, 1), 256, 0, stream>>>(ctx_h, D_, 0, WmhT, D_, 0,
      bmh, 0, qn_f, D_, hhead, nullptr, D_, 0, nullptr, 0, 0,
      B_, D_, D_, 1.f, 0);

  // LN(h_head)
  ln_rows<<<B_, 256, 0, stream>>>(hhead, ln_fc_g, ln_fc_b, hn_h, nullptr);

  // out = qn + relu(LN(h_head) @ W_fc + b_fc)
  gemm16<<<dim3(8, 1), 256, 0, stream>>>(hn_h, D_, 0, WfcT, D_, 0,
      bfc, 0, qn_f, D_, (float*)d_out, nullptr, D_, 0, nullptr, 0, 0,
      B_, D_, D_, 1.f, 1);
}
